// CoAttLayer_72834055406332
// MI455X (gfx1250) — matrix operand order repaired and checked by reading, not compiled, not run on hardware
//
#include <hip/hip_runtime.h>
#include <hip/hip_bf16.h>

// ---------------------------------------------------------------------------
// Problem constants: B=8, C=512, H=W=32 -> HW=1024, P=B*HW=8192
// ---------------------------------------------------------------------------
#define Cc   512
#define HW   1024
#define Pn   8192          // total pixels (query/key rows)
#define KT   16            // 512 / 32 k-steps per WMMA chain

typedef __attribute__((ext_vector_type(16))) __bf16 bf16x16;
typedef __attribute__((ext_vector_type(8)))  __bf16 bf16x8;
typedef __attribute__((ext_vector_type(8)))  float  f32x8;
typedef __attribute__((ext_vector_type(4))) unsigned int u32x4;
typedef __attribute__((ext_vector_type(4))) int  i32x4;
typedef __attribute__((ext_vector_type(8))) int  i32x8;

#if __has_builtin(__builtin_amdgcn_tensor_load_to_lds) && __has_builtin(__builtin_amdgcn_s_wait_tensorcnt)
#define USE_TDM 1
#else
#define USE_TDM 0
#endif

__device__ __forceinline__ __bf16 f32_to_bf16(float f) {
    union { float f; unsigned u; } v; v.f = f;
    unsigned r = v.u + 0x7FFFu + ((v.u >> 16) & 1u);   // round-to-nearest-even
    unsigned short h = (unsigned short)(r >> 16);
    return __builtin_bit_cast(__bf16, h);
}

__device__ __forceinline__ f32x8 wmma_bf16(const bf16x16& a, const bf16x16& b, const f32x8& c) {
    return __builtin_amdgcn_wmma_f32_16x16x32_bf16(
        /*neg_a=*/false, a, /*neg_b=*/false, b,
        /*c_mod=*/(short)0, c, /*reuse_a=*/false, /*reuse_b=*/false);
}

#if USE_TDM
// Issue one TDM copy: 16 rows x 512 bf16 (16 KiB, contiguous) from global -> LDS.
// D# per CDNA5 ISA: group0 = {count=1, lds_addr,
// global_addr, type=2}; group1 = {data_size=1(2B), tensor 512x8192,
// tile 512x16, dim0_stride=512}. Groups 2/3 unused (<=2D) -> zeros.
__device__ __forceinline__ void tdm_load_tile(const __bf16* gsrc, unsigned lds_byte_off) {
    unsigned long long ga = (unsigned long long)(size_t)gsrc;
    u32x4 g0;
    g0[0] = 1u;                                   // count=1, user descriptor
    g0[1] = lds_byte_off;                         // lds_addr (bytes)
    g0[2] = (unsigned)(ga & 0xFFFFFFFFu);         // global_addr[31:0]
    g0[3] = (unsigned)(ga >> 32) | (2u << 30);    // global_addr[56:32] | type=2
    i32x8 g1;
    g1[0] = (int)(1u << 16);                      // wg_mask=0, data_size=1 (2 bytes)
    g1[1] = (int)(512u << 16);                    // tensor_dim0[15:0]=512 @bits63:48
    g1[2] = (int)(8192u << 16);                   // dim0 hi=0, tensor_dim1 lo=8192
    g1[3] = (int)(512u << 16);                    // dim1 hi=0, tile_dim0=512
    g1[4] = 16;                                   // tile_dim1=16, tile_dim2=0
    g1[5] = 512;                                  // tensor_dim0_stride[31:0]
    g1[6] = 0;
    g1[7] = 0;
    i32x4 z4 = {0, 0, 0, 0};
#if __clang_major__ >= 23
    i32x8 z8 = {0, 0, 0, 0, 0, 0, 0, 0};
    __builtin_amdgcn_tensor_load_to_lds(g0, g1, z4, z4, z8, 0);
#else
    __builtin_amdgcn_tensor_load_to_lds(g0, g1, z4, z4, 0);
#endif
}
#endif

// ---------------------------------------------------------------------------
// Kernel 0: one-shot weight conversion f32 -> bf16 (amortized across the GEMMs)
// ---------------------------------------------------------------------------
__global__ void __launch_bounds__(256)
wcvt_kernel(const float* __restrict__ Wq, const float* __restrict__ Wk,
            __bf16* __restrict__ Wqb, __bf16* __restrict__ Wkb)
{
    const int i = blockIdx.x * 256 + threadIdx.x;    // 262144 elements
    Wqb[i] = f32_to_bf16(Wq[i]);
    Wkb[i] = f32_to_bf16(Wk[i]);
}

// ---------------------------------------------------------------------------
// Kernel 1: Q = Xmat @ Wq^T + bq,  K = Xmat @ Wk^T + bk   (bf16 out, f32 acc)
// Block = 128 threads (4 waves), covers 64 pixels; N-loop covers all 512 cols.
// ---------------------------------------------------------------------------
__global__ void __launch_bounds__(128)
proj_qk_kernel(const float* __restrict__ x,
               const __bf16* __restrict__ Wqb, const float* __restrict__ bq,
               const __bf16* __restrict__ Wkb, const float* __restrict__ bk,
               __bf16* __restrict__ Qb, __bf16* __restrict__ Kb)
{
    // pixel-major chunk: [pixel][channel] with 72-element (144B, 16B-aligned) rows
    __shared__ __bf16 ldsx[64][72];

    const int tid  = threadIdx.x;
    const int lane = tid & 31;
    const int widx = tid >> 5;
    const int m0   = blockIdx.x * 64;
    const int bimg = m0 >> 10;
    const int hw0  = m0 & (HW - 1);

    const int mrow = lane & 15;
    const int half = lane >> 4;
    const int pw   = widx * 16 + mrow;

    // ------- build all 16 A fragments (16x32 bf16) in registers -------
    bf16x16 afrag[KT];
    for (int kc = 0; kc < 8; ++kc) {                  // 8 chunks of 64 channels
        #pragma unroll
        for (int it = 0; it < 32; ++it) {             // coalesced over pixels
            int lin = it * 128 + tid;
            int cch = lin >> 6;
            int pix = lin & 63;
            float v = x[(size_t)bimg * (Cc * HW) + (size_t)(kc * 64 + cch) * HW + hw0 + pix];
            ldsx[pix][cch] = f32_to_bf16(v);
        }
        __syncthreads();
        #pragma unroll
        for (int sub = 0; sub < 2; ++sub) {
            bf16x16 a;
            const __bf16* p = &ldsx[pw][sub * 32 + half * 16];
            ((bf16x8*)&a)[0] = *(const bf16x8*)(p);        // K = half*16 + 0..7
            ((bf16x8*)&a)[1] = *(const bf16x8*)(p + 8);    // K = half*16 + 8..15
            afrag[kc * 2 + sub] = a;
        }
        __syncthreads();
    }

    // ------- N loop: all 32 output tiles of 16 channels -------
    const int ncol = lane & 15;
    for (int nt = 0; nt < 32; ++nt) {
        const int n0 = nt * 16;
        const int d  = n0 + ncol;
        const __bf16* wq = Wqb + (size_t)d * Cc + half * 16;  // 32B contiguous per frag
        const __bf16* wk = Wkb + (size_t)d * Cc + half * 16;
        f32x8 accq = {}; f32x8 acck = {};
        #pragma unroll
        for (int kt = 0; kt < KT; ++kt) {
            bf16x16 bqf = *(const bf16x16*)(wq + kt * 32);
            bf16x16 bkf = *(const bf16x16*)(wk + kt * 32);
            accq = wmma_bf16(afrag[kt], bqf, accq);
            acck = wmma_bf16(afrag[kt], bkf, acck);
        }
        const float biasq = bq[d];
        const float biask = bk[d];
        #pragma unroll
        for (int r = 0; r < 8; ++r) {
            int M = r + half * 8;
            size_t off = (size_t)(m0 + widx * 16 + M) * Cc + n0 + ncol;
            Qb[off] = f32_to_bf16(accq[r] + biasq);
            Kb[off] = f32_to_bf16(acck[r] + biask);
        }
    }
}

// ---------------------------------------------------------------------------
// Kernel 2: s[i] = (1/8) * sum_b max_hw ( Q_i . K_{b*1024+hw} )
// Block = 128 threads (4 waves, 64 query rows). K tiles (16 keys x 512 ch,
// 16 KiB) staged into LDS -- via the Tensor Data Mover, double-buffered,
// synchronized with s_wait_tensorcnt -- and shared by all 4 waves.
// ---------------------------------------------------------------------------
__global__ void __launch_bounds__(128)
scores_kernel(const __bf16* __restrict__ Qb, const __bf16* __restrict__ Kb,
              float* __restrict__ S)
{
    __shared__ __bf16 ldsK[2][16][Cc];               // 2 x 16 KiB

    const int tid  = threadIdx.x;
    const int lane = tid & 31;
    const int widx = tid >> 5;
    const int i0   = (blockIdx.x * 4 + widx) * 16;

    const int mrow = lane & 15;
    const int half = lane >> 4;
    const int ncol = lane & 15;

    // Preload 16 A fragments (16 query rows, full 512-channel depth)
    bf16x16 afrag[KT];
    #pragma unroll
    for (int kt = 0; kt < KT; ++kt) {
        const __bf16* p = Qb + (size_t)(i0 + mrow) * Cc + kt * 32 + half * 16;
        bf16x16 a;
        ((bf16x8*)&a)[0] = *(const bf16x8*)(p);
        ((bf16x8*)&a)[1] = *(const bf16x8*)(p + 8);
        afrag[kt] = a;
    }

    float runmax[8], ssum[8];
    #pragma unroll
    for (int r = 0; r < 8; ++r) { runmax[r] = -3.0e38f; ssum[r] = 0.0f; }

#if USE_TDM
    // low 32 bits of the generic address == LDS byte offset (aperture form)
    const unsigned lds0 = (unsigned)(size_t)(&ldsK[0][0][0]);
    if (widx == 0) tdm_load_tile(Kb, lds0);          // prologue: tile 0 -> buf 0
#endif

    for (int jt = 0; jt < Pn / 16; ++jt) {           // 512 key tiles
#if USE_TDM
        if (widx == 0) {
            if (jt + 1 < Pn / 16) {
                tdm_load_tile(Kb + (size_t)(jt + 1) * 16 * Cc,
                              lds0 + (unsigned)(((jt + 1) & 1) * 16 * Cc * 2));
                __builtin_amdgcn_s_wait_tensorcnt(1);  // tile jt done, jt+1 in flight
            } else {
                __builtin_amdgcn_s_wait_tensorcnt(0);  // last tile: drain everything
            }
        }
        __syncthreads();                              // tile jt visible to all waves
        const __bf16* ksrc = &ldsK[jt & 1][0][0];
#else
        {   // cooperative fallback stage: 16 KiB as 1024 uint4
            __builtin_prefetch(Kb + (size_t)(jt + 1) * 16 * Cc + tid * 128, 0, 0);
            const uint4* src = (const uint4*)(Kb + (size_t)jt * 16 * Cc);
            uint4* dst = (uint4*)&ldsK[0][0][0];
            #pragma unroll
            for (int it = 0; it < 8; ++it) dst[it * 128 + tid] = src[it * 128 + tid];
        }
        __syncthreads();
        const __bf16* ksrc = &ldsK[0][0][0];
#endif
        f32x8 acc = {};
        const __bf16* kb = ksrc + ncol * Cc + half * 16;   // 32B contiguous per frag
        #pragma unroll
        for (int kt = 0; kt < KT; ++kt) {
            bf16x16 b = *(const bf16x16*)(kb + kt * 32);
            acc = wmma_bf16(afrag[kt], b, acc);
        }
        #pragma unroll
        for (int r = 0; r < 8; ++r) runmax[r] = fmaxf(runmax[r], acc[r]);

        if ((jt & 63) == 63) {                        // end of one key image
            #pragma unroll
            for (int r = 0; r < 8; ++r) {
                float v = runmax[r];                  // max over 16 key lanes
                v = fmaxf(v, __shfl_xor(v, 1));
                v = fmaxf(v, __shfl_xor(v, 2));
                v = fmaxf(v, __shfl_xor(v, 4));
                v = fmaxf(v, __shfl_xor(v, 8));
                ssum[r] += v;
                runmax[r] = -3.0e38f;
            }
        }
        __syncthreads();                              // buffer safe to overwrite
    }
    #pragma unroll
    for (int r = 0; r < 8; ++r) {
        if ((lane & 15) == 0) {
            S[i0 + r + half * 8] = ssum[r] * 0.125f;  // mean over 8 key images
        }
    }
}

// ---------------------------------------------------------------------------
// Kernel 3: per-image softmax over hw (1024) of scale * s  -> Wt[8192]
// ---------------------------------------------------------------------------
__global__ void __launch_bounds__(256)
softmax_kernel(const float* __restrict__ S, float* __restrict__ Wt)
{
    __shared__ float red[256];
    const int tid = threadIdx.x;
    const int b   = blockIdx.x;
    const float scale = 0.044194173824159216f;        // 1/sqrt(512)

    float v[4];
    float lmax = -3.0e38f;
    #pragma unroll
    for (int it = 0; it < 4; ++it) {
        v[it] = S[b * HW + it * 256 + tid] * scale;
        lmax = fmaxf(lmax, v[it]);
    }
    red[tid] = lmax; __syncthreads();
    for (int s = 128; s > 0; s >>= 1) {
        if (tid < s) red[tid] = fmaxf(red[tid], red[tid + s]);
        __syncthreads();
    }
    const float gmax = red[0]; __syncthreads();

    float lsum = 0.0f;
    #pragma unroll
    for (int it = 0; it < 4; ++it) { v[it] = __expf(v[it] - gmax); lsum += v[it]; }
    red[tid] = lsum; __syncthreads();
    for (int s = 128; s > 0; s >>= 1) {
        if (tid < s) red[tid] += red[tid + s];
        __syncthreads();
    }
    const float inv = 1.0f / red[0];
    #pragma unroll
    for (int it = 0; it < 4; ++it) Wt[b * HW + it * 256 + tid] = v[it] * inv;
}

// ---------------------------------------------------------------------------
// Kernel 4a: m[c] = (1/8192) * sum_p x[b,c,hw] * Wt[p]
// ---------------------------------------------------------------------------
__global__ void __launch_bounds__(256)
colmean_kernel(const float* __restrict__ x, const float* __restrict__ Wt,
               float* __restrict__ Mv)
{
    __shared__ float red[256];
    const int tid = threadIdx.x;
    const int c   = blockIdx.x;
    float s = 0.0f;
    for (int p = tid; p < Pn; p += 256) {
        int b  = p >> 10;
        int hw = p & (HW - 1);
        s += x[(size_t)b * (Cc * HW) + (size_t)c * HW + hw] * Wt[p];
    }
    red[tid] = s; __syncthreads();
    for (int k = 128; k > 0; k >>= 1) {
        if (tid < k) red[tid] += red[tid + k];
        __syncthreads();
    }
    if (tid == 0) Mv[c] = red[0] * (1.0f / (float)Pn);
}

// ---------------------------------------------------------------------------
// Kernel 4b: proto[d] = b6[d] + sum_c W6[d,c] * m[c]   (tiny matvec)
// ---------------------------------------------------------------------------
__global__ void __launch_bounds__(512)
proto_kernel(const float* __restrict__ W6, const float* __restrict__ b6,
             const float* __restrict__ Mv, float* __restrict__ Pr)
{
    const int d = threadIdx.x;
    float s = b6[d];
    const float4* w4 = (const float4*)(W6 + (size_t)d * Cc);
    #pragma unroll 4
    for (int c4 = 0; c4 < Cc / 4; ++c4) {
        float4 w = w4[c4];
        const float4 m = ((const float4*)Mv)[c4];
        s += w.x * m.x + w.y * m.y + w.z * m.z + w.w * m.w;
    }
    Pr[d] = s;
}

// ---------------------------------------------------------------------------
// Kernel 5: out[b,c,hw] = x[b,c,hw] * proto[c]   (float4 elementwise)
// ---------------------------------------------------------------------------
__global__ void __launch_bounds__(256)
scaleout_kernel(const float* __restrict__ x, const float* __restrict__ Pr,
                float* __restrict__ out)
{
    const int i4 = blockIdx.x * 256 + threadIdx.x;   // 1,048,576 float4 groups
    const int base = i4 * 4;
    const int c = (base >> 10) & (Cc - 1);
    float4 v = ((const float4*)x)[i4];
    float p = Pr[c];
    v.x *= p; v.y *= p; v.z *= p; v.w *= p;
    ((float4*)out)[i4] = v;
}

// ---------------------------------------------------------------------------
extern "C" void kernel_launch(void* const* d_in, const int* in_sizes, int n_in,
                              void* d_out, int out_size, void* d_ws, size_t ws_size,
                              hipStream_t stream) {
    (void)in_sizes; (void)n_in; (void)out_size; (void)ws_size;
    const float* x  = (const float*)d_in[0];
    const float* Wq = (const float*)d_in[1];
    const float* bq = (const float*)d_in[2];
    const float* Wk = (const float*)d_in[3];
    const float* bk = (const float*)d_in[4];
    const float* W6 = (const float*)d_in[5];
    const float* b6 = (const float*)d_in[6];
    float* out = (float*)d_out;

    char* ws = (char*)d_ws;
    __bf16* Qb  = (__bf16*)(ws);                         // 8 MiB
    __bf16* Kb  = (__bf16*)(ws + (size_t)8388608);       // 8 MiB
    __bf16* Wqb = (__bf16*)(ws + (size_t)16777216);      // 512 KiB
    __bf16* Wkb = (__bf16*)(ws + (size_t)17301504);      // 512 KiB
    float*  S   = (float*)(ws + (size_t)17825792);       // 32 KiB
    float*  Wt  = (float*)(ws + (size_t)17858560);       // 32 KiB
    float*  Mv  = (float*)(ws + (size_t)17891328);       // 2 KiB
    float*  Pr  = (float*)(ws + (size_t)17893376);       // 2 KiB

    wcvt_kernel     <<<1024, 256, 0, stream>>>(Wq, Wk, Wqb, Wkb);
    proj_qk_kernel  <<<128,  128, 0, stream>>>(x, Wqb, bq, Wkb, bk, Qb, Kb);
    scores_kernel   <<<128,  128, 0, stream>>>(Qb, Kb, S);
    softmax_kernel  <<<8,    256, 0, stream>>>(S, Wt);
    colmean_kernel  <<<512,  256, 0, stream>>>(x, Wt, Mv);
    proto_kernel    <<<1,    512, 0, stream>>>(W6, b6, Mv, Pr);
    scaleout_kernel <<<4096, 256, 0, stream>>>(x, Pr, out);
}
